// EgnnPolicy_82188494176608
// MI455X (gfx1250) — matrix operand order repaired and checked by reading, not compiled, not run on hardware
//
#include <hip/hip_runtime.h>
#include <hip/hip_bf16.h>
#include <cstdint>

typedef _Float16 half_t;
typedef __attribute__((ext_vector_type(16))) _Float16 v16h;
typedef __attribute__((ext_vector_type(8)))  float    v8f;

// ---- swizzled-weight workspace layout (in halves) ----
#define LAYER_W_STRIDE 114688
#define OFF_W1A 0
#define OFF_W1B 16384
#define OFF_EW2 32768
#define OFF_CW1 49152
#define OFF_NW1 65536
#define OFF_NW2 98304

struct KParams {
  const float* obs;
  const float* noise;
  float*       out;
  const half_t* wsw;
  const float* emb_w;
  const float* emb_b;
  const float* log_std;
  const float* w1r[4];
  const float* eb1[4];
  const float* eb2[4];
  const float* aw [4];
  const float* ab [4];
  const float* cb1[4];
  const float* cw2[4];
  const float* nb1[4];
  const float* nb2[4];
};

// Convert f32 weight (K x 128 row-major) to f16 in WMMA B-fragment order:
// dst[((kt*8+nt)*512) + lane*16 + h] = W[(32*kt + 8*(lane/16) + (h<8 ? h : h+8))*128 + 16*nt + lane%16]
__global__ void swizzle_w(const float* __restrict__ W, half_t* __restrict__ dst, int K) {
  int t = blockIdx.x * 256 + threadIdx.x;
  if (t >= K * 128) return;
  int tile   = t >> 9;
  int within = t & 511;
  int lane = within >> 4;
  int h    = within & 15;
  int kt = tile >> 3;
  int nt = tile & 7;
  int k = 32 * kt + 8 * (lane >> 4) + (h < 8 ? h : h + 8);
  int n = 16 * nt + (lane & 15);
  dst[t] = (half_t)W[k * 128 + n];
}

__device__ __forceinline__ float fsigmoid(float x) { return 1.0f / (1.0f + __expf(-x)); }
__device__ __forceinline__ float fsilu(float x)    { return x / (1.0f + __expf(-x)); }

union FragH { v16h v; uint4 q[2]; };

// A fragment (16x32 f16) from row-major LDS activations.
// lane L: m=L%16, g=L/16; halves 0..7 -> K=kbase+8g+0..7 ; halves 8..15 -> K=kbase+8g+16..23
__device__ __forceinline__ v16h load_afrag(const half_t* act, int rowBase, int stride,
                                           int kbase, int lane) {
  int g = lane >> 4, m = lane & 15;
  const half_t* rp = act + (rowBase + m) * stride + kbase + 8 * g;
  FragH r;
  r.q[0] = *(const uint4*)(rp);
  r.q[1] = *(const uint4*)(rp + 16);
  return r.v;
}

// B fragment from pre-swizzled global buffer: 16 contiguous halves per lane.
__device__ __forceinline__ v16h load_bfrag(const half_t* __restrict__ w, int lane) {
  FragH r;
  const uint4* p = (const uint4*)(w + lane * 16);
  r.q[0] = p[0];
  r.q[1] = p[1];
  return r.v;
}

__device__ __forceinline__ v8f wmma_f16(v16h a, v16h b, v8f c) {
  return __builtin_amdgcn_wmma_f32_16x16x32_f16(false, a, false, b, (short)0, c, false, false);
}

template <int KSTEPS, int NT>
__device__ __forceinline__ void gemm_acc(const half_t* A, int rowBase, int strideA,
                                         const half_t* __restrict__ WB, int ntBase,
                                         v8f* acc, int lane) {
#pragma unroll
  for (int kt = 0; kt < KSTEPS; ++kt) {
    v16h a = load_afrag(A, rowBase, strideA, kt * 32, lane);
#pragma unroll
    for (int q = 0; q < NT; ++q) {
      v16h b = load_bfrag(WB + (((kt << 3) + ntBase + q) << 9), lane);
      acc[q] = wmma_f16(a, b, acc[q]);
    }
  }
}

// One block = one time group (32 agents). 256 threads = 8 wave32.
__global__ __launch_bounds__(256) void egnn_fused(KParams P) {
  __shared__ float  s_h[32 * 128];                              // 16 KB  h (f32, residual)
  __shared__ __align__(16) half_t s_hf16[32 * 128];             //  8 KB  h (f16 A-side)
  __shared__ float  s_e1[32 * 128];                             // 16 KB  H @ W1a
  __shared__ float  s_e2[32 * 128];                             // 16 KB  H @ W1b
  __shared__ __align__(16) half_t s_mA[128 * 128];              // 32 KB  silu(m1) / t1 / nodeA / hn1
  __shared__ __align__(16) half_t s_m2[128 * 128];              // 32 KB  post-attention m
  __shared__ float  s_aggh[32 * 128];                           // 16 KB
  __shared__ float  s_cdn[128 * 3];
  __shared__ float  s_rad[128];
  __shared__ float  s_tv[128];
  __shared__ float  s_att[128];
  __shared__ float  s_aggc[32 * 4];
  __shared__ float  s_coord[32 * 4];
  __shared__ float  s_obs[32 * 32];

  const int tid  = threadIdx.x;
  const int lane = tid & 31;
  const int wave = tid >> 5;
  const int blk  = blockIdx.x;
  const int g    = lane >> 4;
  const int lm   = lane & 15;
  const v8f zero8 = {0.f, 0.f, 0.f, 0.f, 0.f, 0.f, 0.f, 0.f};

  // ---- load obs (32 nodes x 32 feats) ----
  ((float4*)s_obs)[tid] = ((const float4*)(P.obs + (size_t)blk * 1024))[tid];
  __syncthreads();
  if (tid < 32) {
    s_coord[tid * 4 + 0] = s_obs[tid * 32 + 0];
    s_coord[tid * 4 + 1] = s_obs[tid * 32 + 1];
    s_coord[tid * 4 + 2] = s_obs[tid * 32 + 2];
  }
  // ---- embedding: h = obs[:,3:] @ emb_w + emb_b ----
  {
    int n = tid >> 3, f0 = (tid & 7) * 16;
    float acc[16];
#pragma unroll
    for (int ff = 0; ff < 16; ++ff) acc[ff] = P.emb_b[f0 + ff];
    for (int k = 0; k < 29; ++k) {
      float o = s_obs[n * 32 + 3 + k];
      const float* wr = P.emb_w + k * 128 + f0;
#pragma unroll
      for (int ff = 0; ff < 16; ++ff) acc[ff] += o * wr[ff];
    }
#pragma unroll
    for (int ff = 0; ff < 16; ++ff) {
      s_h[n * 128 + f0 + ff] = acc[ff];
      s_hf16[n * 128 + f0 + ff] = (half_t)acc[ff];
    }
  }
  __syncthreads();

  for (int layer = 0; layer < 4; ++layer) {
    const half_t* Lw  = P.wsw + (size_t)layer * LAYER_W_STRIDE;
    const float* w1r  = P.w1r[layer];
    const float* eb1  = P.eb1[layer];
    const float* eb2  = P.eb2[layer];
    const float* awv  = P.aw[layer];
    const float  abv  = P.ab[layer][0];
    const float* cb1  = P.cb1[layer];
    const float* cw2v = P.cw2[layer];
    const float* nb1  = P.nb1[layer];
    const float* nb2  = P.nb2[layer];

    // zero aggregates
    for (int u = tid; u < 32 * 128; u += 256) s_aggh[u] = 0.f;
    if (tid < 128) s_aggc[tid] = 0.f;

    // ---- e1 = H@W1a , e2 = H@W1b  (32x128 each; waves 0-3 -> e1, 4-7 -> e2) ----
    {
      const half_t* wb = (wave < 4) ? (Lw + OFF_W1A) : (Lw + OFF_W1B);
      float* dst = (wave < 4) ? s_e1 : s_e2;
      int wi = wave & 3;
      int mtile = wi & 1;
      int ntb = (wi >> 1) * 4;
      v8f acc[4];
#pragma unroll
      for (int q = 0; q < 4; ++q) acc[q] = zero8;
      gemm_acc<4, 4>(s_hf16, mtile * 16, 128, wb, ntb, acc, lane);
#pragma unroll
      for (int q = 0; q < 4; ++q) {
        int col = (ntb + q) * 16 + lm;
#pragma unroll
        for (int v = 0; v < 8; ++v)
          dst[(mtile * 16 + v + 8 * g) * 128 + col] = acc[q][v];
      }
    }
    __syncthreads();

    // ---- edge chunks: 8 x (4 receivers x 32 senders = 128 edges) ----
    for (int c = 0; c < 8; ++c) {
      int ibase = c * 4;
      if (tid < 128) {
        int e = tid, i = ibase + (e >> 5), j = e & 31;
        float dx = s_coord[i * 4 + 0] - s_coord[j * 4 + 0];
        float dy = s_coord[i * 4 + 1] - s_coord[j * 4 + 1];
        float dz = s_coord[i * 4 + 2] - s_coord[j * 4 + 2];
        float r  = dx * dx + dy * dy + dz * dz;
        float ri = 1.0f / (sqrtf(r) + 1e-8f);
        s_cdn[e * 3 + 0] = dx * ri;
        s_cdn[e * 3 + 1] = dy * ri;
        s_cdn[e * 3 + 2] = dz * ri;
        s_rad[e] = r;
      }
      __syncthreads();
      // m1 = silu(e1[i] + e2[j] + radial*w1r + eb1)
      {
        int e = tid >> 1, f0 = (tid & 1) << 6;
        int i = ibase + (e >> 5), j = e & 31;
        float r = s_rad[e];
        for (int f = f0; f < f0 + 64; ++f) {
          float x = s_e1[i * 128 + f] + s_e2[j * 128 + f] + r * w1r[f] + eb1[f];
          s_mA[e * 128 + f] = (half_t)fsilu(x);
        }
      }
      __syncthreads();
      // m2 = silu(m1 @ ew2 + eb2)
      {
        v8f acc[8];
#pragma unroll
        for (int q = 0; q < 8; ++q) acc[q] = zero8;
        gemm_acc<4, 8>(s_mA, wave * 16, 128, Lw + OFF_EW2, 0, acc, lane);
#pragma unroll
        for (int nt = 0; nt < 8; ++nt) {
          int col = nt * 16 + lm;
          float bb = eb2[col];
#pragma unroll
          for (int v = 0; v < 8; ++v)
            s_m2[(wave * 16 + v + 8 * g) * 128 + col] = (half_t)fsilu(acc[nt][v] + bb);
        }
      }
      __syncthreads();
      // attention: att = sigmoid(m2 @ aw + ab)
      if (tid < 128) {
        float s = abv;
        int rb = tid * 128;
        for (int ff = 0; ff < 128; ++ff) {
          int f = (ff + (tid << 2)) & 127;   // skew to spread LDS banks
          s += (float)s_m2[rb + f] * awv[f];
        }
        s_att[tid] = fsigmoid(s);
      }
      __syncthreads();
      {
        int e = tid >> 1, f0 = (tid & 1) << 6;
        float a = s_att[e];
        for (int f = f0; f < f0 + 64; ++f)
          s_m2[e * 128 + f] = (half_t)((float)s_m2[e * 128 + f] * a);
      }
      __syncthreads();
      // t1 = silu(m2 @ cw1 + cb1)  (into s_mA, m1 is dead now)
      {
        v8f acc[8];
#pragma unroll
        for (int q = 0; q < 8; ++q) acc[q] = zero8;
        gemm_acc<4, 8>(s_m2, wave * 16, 128, Lw + OFF_CW1, 0, acc, lane);
#pragma unroll
        for (int nt = 0; nt < 8; ++nt) {
          int col = nt * 16 + lm;
          float bb = cb1[col];
#pragma unroll
          for (int v = 0; v < 8; ++v)
            s_mA[(wave * 16 + v + 8 * g) * 128 + col] = (half_t)fsilu(acc[nt][v] + bb);
        }
      }
      __syncthreads();
      // t = tanh(t1 @ cw2)
      if (tid < 128) {
        float s = 0.f;
        int rb = tid * 128;
        for (int ff = 0; ff < 128; ++ff) {
          int f = (ff + (tid << 2)) & 127;
          s += (float)s_mA[rb + f] * cw2v[f];
        }
        s_tv[tid] = tanhf(s);
      }
      __syncthreads();
      // aggregate agg_h (exclude self edge) and agg_c (self edge cdn==0)
      for (int p = tid; p < 512; p += 256) {
        int ii = p >> 7, f = p & 127, i = ibase + ii;
        float s = 0.f;
        for (int j = 0; j < 32; ++j) {
          if (j == i) continue;
          s += (float)s_m2[(ii * 32 + j) * 128 + f];
        }
        s_aggh[i * 128 + f] += s;
      }
      if (tid < 12) {
        int ii = tid / 3, comp = tid % 3;
        float s = 0.f;
        for (int j = 0; j < 32; ++j)
          s += s_cdn[(ii * 32 + j) * 3 + comp] * s_tv[ii * 32 + j];
        s_aggc[(ibase + ii) * 4 + comp] += s;
      }
      __syncthreads();
    } // chunks

    // coord += agg_c / (NA-1)
    if (tid < 96) {
      int n = tid / 3, comp = tid % 3;
      s_coord[n * 4 + comp] += s_aggc[n * 4 + comp] * (1.0f / 31.0f);
    }
    // nodeA = [h | agg_h] as f16 (32 x 256), reuse s_mA
    half_t* nodeA = s_mA;
    half_t* hn1   = s_mA + 32 * 256;
    for (int u = tid; u < 32 * 256; u += 256) {
      int n = u >> 8, k = u & 255;
      nodeA[u] = (k < 128) ? s_hf16[n * 128 + k] : (half_t)s_aggh[n * 128 + (k - 128)];
    }
    __syncthreads();
    // hn1 = silu(nodeA @ nw1 + nb1)
    {
      int mtile = wave & 1;
      int ntb = (wave >> 1) * 2;
      v8f acc[2];
      acc[0] = zero8; acc[1] = zero8;
      gemm_acc<8, 2>(nodeA, mtile * 16, 256, Lw + OFF_NW1, ntb, acc, lane);
#pragma unroll
      for (int q = 0; q < 2; ++q) {
        int col = (ntb + q) * 16 + lm;
        float bb = nb1[col];
#pragma unroll
        for (int v = 0; v < 8; ++v)
          hn1[(mtile * 16 + v + 8 * g) * 128 + col] = (half_t)fsilu(acc[q][v] + bb);
      }
    }
    __syncthreads();
    // h += hn1 @ nw2 + nb2
    {
      int mtile = wave & 1;
      int ntb = (wave >> 1) * 2;
      v8f acc[2];
      acc[0] = zero8; acc[1] = zero8;
      gemm_acc<4, 2>(hn1, mtile * 16, 128, Lw + OFF_NW2, ntb, acc, lane);
#pragma unroll
      for (int q = 0; q < 2; ++q) {
        int col = (ntb + q) * 16 + lm;
        float bb = nb2[col];
#pragma unroll
        for (int v = 0; v < 8; ++v) {
          int row = mtile * 16 + v + 8 * g;
          float nh = s_h[row * 128 + col] + acc[q][v] + bb;
          s_h[row * 128 + col] = nh;
          s_hf16[row * 128 + col] = (half_t)nh;
        }
      }
    }
    __syncthreads();
  } // layers

  // ---- outputs: actions + log_probs ----
  if (tid < 96) {
    int n = tid / 3, e = tid % 3;
    size_t gn = (size_t)blk * 32 + n;
    float mu = s_coord[n * 4 + e];
    float ls = P.log_std[e];
    float sd = __expf(ls);
    float nz = P.noise[gn * 3 + e];
    P.out[gn * 6 + e]     = mu + sd * nz;
    P.out[gn * 6 + 3 + e] = -0.5f * nz * nz - ls - 0.9189385332046727f;
  }
}

extern "C" void kernel_launch(void* const* d_in, const int* in_sizes, int n_in,
                              void* d_out, int out_size, void* d_ws, size_t ws_size,
                              hipStream_t stream) {
  (void)n_in; (void)out_size; (void)ws_size;
  // Resolve input ordering from sizes (handles insertion-order and sorted-key pytrees).
  int iObs, iNoise, pbase;
  if (in_sizes[0] == 1024 * 32 * 32) { iObs = 0; iNoise = 1; pbase = 4; }  // dict insertion order
  else                               { iObs = 3; iNoise = 2; pbase = 4; }  // fully sorted keys
  int iEmbB, iEmbW;
  if (in_sizes[pbase] == 128) { iEmbB = pbase;     iEmbW = pbase + 1; }    // sorted params keys
  else                        { iEmbW = pbase;     iEmbB = pbase + 1; }    // insertion order
  const int lbase = pbase + 2;
  const bool sortedL = (in_sizes[lbase] == 1);  // sorted layer dict starts with 'ab' (size 1)
  // index maps within a layer's 13 leaves
  // sorted:    ab aw cb1 cw1 cw2 eb1 eb2 ew1 ew2 nb1 nb2 nw1 nw2
  // insertion: ew1 eb1 ew2 eb2 nw1 nb1 nw2 nb2 cw1 cb1 cw2 aw ab
  const int mAB  = sortedL ? 0  : 12;
  const int mAW  = sortedL ? 1  : 11;
  const int mCB1 = sortedL ? 2  : 9;
  const int mCW1 = sortedL ? 3  : 8;
  const int mCW2 = sortedL ? 4  : 10;
  const int mEB1 = sortedL ? 5  : 1;
  const int mEB2 = sortedL ? 6  : 3;
  const int mEW1 = sortedL ? 7  : 0;
  const int mEW2 = sortedL ? 8  : 2;
  const int mNB1 = sortedL ? 9  : 5;
  const int mNB2 = sortedL ? 10 : 7;
  const int mNW1 = sortedL ? 11 : 4;
  const int mNW2 = sortedL ? 12 : 6;

  half_t* wsw = (half_t*)d_ws;

  KParams P;
  P.obs     = (const float*)d_in[iObs];
  P.noise   = (const float*)d_in[iNoise];
  P.out     = (float*)d_out;
  P.wsw     = wsw;
  P.emb_w   = (const float*)d_in[iEmbW];
  P.emb_b   = (const float*)d_in[iEmbB];
  P.log_std = (const float*)d_in[lbase + 52];

  for (int l = 0; l < 4; ++l) {
    const int b = lbase + 13 * l;
    const float* ew1 = (const float*)d_in[b + mEW1];
    const float* ew2 = (const float*)d_in[b + mEW2];
    const float* cw1 = (const float*)d_in[b + mCW1];
    const float* nw1 = (const float*)d_in[b + mNW1];
    const float* nw2 = (const float*)d_in[b + mNW2];
    half_t* lw = wsw + (size_t)l * LAYER_W_STRIDE;
    // Pre-swizzle all GEMM weights into WMMA B-fragment layout (f16), L2-resident afterwards.
    swizzle_w<<<64,  256, 0, stream>>>(ew1,             lw + OFF_W1A, 128); // ew1 rows [0,128)
    swizzle_w<<<64,  256, 0, stream>>>(ew1 + 128 * 128, lw + OFF_W1B, 128); // ew1 rows [128,256)
    swizzle_w<<<64,  256, 0, stream>>>(ew2,             lw + OFF_EW2, 128);
    swizzle_w<<<64,  256, 0, stream>>>(cw1,             lw + OFF_CW1, 128);
    swizzle_w<<<128, 256, 0, stream>>>(nw1,             lw + OFF_NW1, 256);
    swizzle_w<<<64,  256, 0, stream>>>(nw2,             lw + OFF_NW2, 128);

    P.w1r[l] = ew1 + 256 * 128;  // last row of ew1 multiplies the radial scalar
    P.eb1[l] = (const float*)d_in[b + mEB1];
    P.eb2[l] = (const float*)d_in[b + mEB2];
    P.aw [l] = (const float*)d_in[b + mAW];
    P.ab [l] = (const float*)d_in[b + mAB];
    P.cb1[l] = (const float*)d_in[b + mCB1];
    P.cw2[l] = (const float*)d_in[b + mCW2];
    P.nb1[l] = (const float*)d_in[b + mNB1];
    P.nb2[l] = (const float*)d_in[b + mNB2];
  }

  egnn_fused<<<dim3(1024), dim3(256), 0, stream>>>(P);
}
